// MultiHeadAttention_6133213298798
// MI455X (gfx1250) — matrix operand order repaired and checked by reading, not compiled, not run on hardware
//
#include <hip/hip_runtime.h>
#include <hip/hip_bf16.h>

// ---------------------------------------------------------------------------
// MI455X (gfx1250) multi-head attention, bf16 WMMA + async global->LDS:
//   0) f32 -> bf16 conversion of X and all weights (one pass, ws-resident)
//   1) fused QKV projection  : bf16 GEMM, 64x64 wave tiles, double-buffered
//                              async LDS staging (peeled steady-state loop)
//   2) flash attention       : bf16 Q,K,V -> bf16 attn, async K/Q staging
//   3) output projection     : bf16 GEMM -> f32 out, same GEMM structure
// Matmuls: v_wmma_f32_16x16x32_bf16.  Staging: global_load_async_to_lds_b128
// (ASYNCcnt) with s_wait_asynccnt; LDS RAW via s_wait_dscnt.
// ---------------------------------------------------------------------------

typedef __attribute__((ext_vector_type(16))) __bf16 bf16x16;
typedef __attribute__((ext_vector_type(8)))  __bf16 bf16x8;
typedef __attribute__((ext_vector_type(8)))  float  f32x8;

#define D_MODEL 512
#define N_HEADS 8
#define D_HEAD  64
#define SEQ_L   4096
#define BATCH   2

// ---- CDNA5 async copy helpers --------------------------------------------
__device__ __forceinline__ void async_load_b128(void* lds_dst, const void* gsrc) {
    unsigned l = (unsigned)(unsigned long long)(uintptr_t)lds_dst;  // LDS offset
    asm volatile("global_load_async_to_lds_b128 %0, %1, off"
                 :: "v"(l), "v"(gsrc)
                 : "memory");
}
__device__ __forceinline__ void wait_async0() {
    asm volatile("s_wait_asynccnt 0" ::: "memory");
}
__device__ __forceinline__ void wait_ds0() {
    asm volatile("s_wait_dscnt 0" ::: "memory");
}

// ---- WMMA fragment helpers (layouts per CDNA5 ISA 7.12.2, wave32) ---------
// A (16x32 bf16): lane = (row r = lane&15, half = lane>>4)
//   elements 0..7  = A[r, half*8 + t], elements 8..15 = A[r, 16 + half*8 + t]
__device__ __forceinline__ bf16x16 frag_a(const __bf16* row, int half) {
    bf16x8 lo = *(const bf16x8*)(row + half * 8);
    bf16x8 hi = *(const bf16x8*)(row + 16 + half * 8);
    bf16x16 r;
#pragma unroll
    for (int i = 0; i < 8; ++i) { r[i] = lo[i]; r[i + 8] = hi[i]; }
    return r;
}
// B (32x16 bf16): lane = (col n = lane&15, half = lane>>4)
//   elements 0..7 = B[half*8 + t, n], elements 8..15 = B[16 + half*8 + t, n]; B rows (fixed n) are contiguous in LDS.
__device__ __forceinline__ bf16x16 frag_b(const __bf16* row, int half) {
    bf16x8 lo = *(const bf16x8*)(row + half * 8);
    bf16x8 hi = *(const bf16x8*)(row + 16 + half * 8);
    bf16x16 r;
#pragma unroll
    for (int i = 0; i < 8; ++i) { r[i] = lo[i]; r[i + 8] = hi[i]; }
    return r;
}
__device__ __forceinline__ f32x8 wmma_bf16(bf16x16 a, bf16x16 b, f32x8 c) {
    return __builtin_amdgcn_wmma_f32_16x16x32_bf16(
        false, a, false, b, (short)0, c, false, false);
}

// ---------------------------------------------------------------------------
// Kernel 0: f32 -> bf16 elementwise conversion (8 elems / thread).
// ---------------------------------------------------------------------------
__global__ __launch_bounds__(256)
void mha_cvt_bf16_kernel(const float* __restrict__ src,
                         __bf16* __restrict__ dst) {
    size_t i = ((size_t)blockIdx.x * 256 + threadIdx.x) * 8;
    float4 a = *(const float4*)(src + i);
    float4 b = *(const float4*)(src + i + 4);
    const float* f = (const float*)&a;
    bf16x8 o;
#pragma unroll
    for (int e = 0; e < 4; ++e) o[e] = (__bf16)f[e];
    const float* g = (const float*)&b;
#pragma unroll
    for (int e = 0; e < 4; ++e) o[e + 4] = (__bf16)g[e];
    *(bf16x8*)(dst + i) = o;
}

// ---------------------------------------------------------------------------
// GEMM core shared by kernels 1 and 3:
//   block = 256 threads (8 waves, 2x4), block tile 128(M) x 256(N),
//   wave tile 64x64 (4x4 WMMA accumulators), K step 32, double-buffered
//   async global->LDS staging with peeled last iteration.
// ---------------------------------------------------------------------------
#define GEMM_K D_MODEL

// ---------------------------------------------------------------------------
// Kernel 1: fused QKV projection.  C[m,n] = sum_k X[m,k] * W[n,k]  (all bf16)
// Output scattered to [B,H,L,dh] bf16 for the attention kernel.
// ---------------------------------------------------------------------------
__global__ __launch_bounds__(256)
void mha_qkv_proj_kernel(const __bf16* __restrict__ X,
                         const __bf16* __restrict__ Wq,
                         const __bf16* __restrict__ Wk,
                         const __bf16* __restrict__ Wv,
                         __bf16* __restrict__ Qo,
                         __bf16* __restrict__ Ko,
                         __bf16* __restrict__ Vo) {
    const __bf16* W   = (blockIdx.z == 0) ? Wq : (blockIdx.z == 1) ? Wk : Wv;
    __bf16*       Dst = (blockIdx.z == 0) ? Qo : (blockIdx.z == 1) ? Ko : Vo;

    __shared__ __align__(16) __bf16 sA[2][128][40];  // 32 K + 8 pad
    __shared__ __align__(16) __bf16 sB[2][256][40];

    const int t    = threadIdx.x;
    const int arow = t >> 1;            // A staging row 0..127
    const int kseg = (t & 1) << 4;      // A element offset 0 or 16
    const int wv   = t >> 5;
    const int wm   = wv >> 2;           // 0..1  (64-row slab)
    const int wn   = wv & 3;            // 0..3  (64-col slab of 256)
    const int lane = t & 31;
    const int half = lane >> 4;
    const int ll   = lane & 15;

    const size_t mbase = (size_t)blockIdx.x * 128;
    const int    nbase = blockIdx.y * 256;

    f32x8 acc[4][4];
#pragma unroll
    for (int i = 0; i < 4; ++i)
#pragma unroll
        for (int j = 0; j < 4; ++j)
#pragma unroll
            for (int e = 0; e < 8; ++e) acc[i][j][e] = 0.0f;

    const __bf16* gA = X + (mbase + arow) * GEMM_K + kseg;   // 32B / thread
    const __bf16* gB = W + (size_t)(nbase + t) * GEMM_K;     // 64B / thread

    auto stage = [&](int buf, int k0) {
        const __bf16* pA = gA + k0;
        async_load_b128(&sA[buf][arow][kseg],     pA);
        async_load_b128(&sA[buf][arow][kseg + 8], pA + 8);
        const __bf16* pB = gB + k0;
        async_load_b128(&sB[buf][t][0],  pB);
        async_load_b128(&sB[buf][t][8],  pB + 8);
        async_load_b128(&sB[buf][t][16], pB + 16);
        async_load_b128(&sB[buf][t][24], pB + 24);
    };
    auto compute = [&](int buf) {
        bf16x16 af[4];
#pragma unroll
        for (int i = 0; i < 4; ++i)
            af[i] = frag_a(&sA[buf][wm * 64 + i * 16 + ll][0], half);
#pragma unroll
        for (int j = 0; j < 4; ++j) {
            bf16x16 bfr = frag_b(&sB[buf][wn * 64 + j * 16 + ll][0], half);
#pragma unroll
            for (int i = 0; i < 4; ++i)
                acc[i][j] = wmma_bf16(af[i], bfr, acc[i][j]);
        }
    };

    stage(0, 0);
    int cur = 0;
    for (int k0 = 0; k0 < GEMM_K - 32; k0 += 32) {  // steady state: branch-free
        wait_async0();
        __syncthreads();
        stage(cur ^ 1, k0 + 32);
        compute(cur);
        cur ^= 1;
    }
    wait_async0();
    __syncthreads();
    compute(cur);

    // scatter store -> [B, H, L, dh] bf16
#pragma unroll
    for (int i = 0; i < 4; ++i)
#pragma unroll
        for (int j = 0; j < 4; ++j)
#pragma unroll
            for (int e = 0; e < 8; ++e) {
                int m = (int)mbase + wm * 64 + i * 16 + (half ? 8 + e : e);
                int n = nbase + wn * 64 + j * 16 + ll;
                int b = m >> 12, l = m & (SEQ_L - 1);
                int h = n >> 6,  d = n & (D_HEAD - 1);
                Dst[(((size_t)b * N_HEADS + h) * SEQ_L + l) * D_HEAD + d] =
                    (__bf16)acc[i][j][e];
            }
}

// ---------------------------------------------------------------------------
// Kernel 2: flash attention per (b,h).  128 query rows / block, 8 waves x 16
// rows.  Keys streamed in 128-key LDS chunks (async staged); online softmax.
// ---------------------------------------------------------------------------
__global__ __launch_bounds__(256)
void mha_flash_attn_kernel(const __bf16* __restrict__ Q,
                           const __bf16* __restrict__ K,
                           const __bf16* __restrict__ V,
                           __bf16* __restrict__ O) {
    const int qt = blockIdx.x;
    const int h  = blockIdx.y;
    const int b  = blockIdx.z;
    const size_t hoff = ((size_t)b * N_HEADS + h) * (size_t)SEQ_L * D_HEAD;
    const __bf16* Qh = Q + hoff;
    const __bf16* Kh = K + hoff;
    const __bf16* Vh = V + hoff;

    __shared__ __align__(16) __bf16 sK [128][72];      // K rows, dh contiguous
    __shared__ __align__(16) __bf16 sVt[64][136];      // V transposed: [d][key]
    __shared__ __align__(16) __bf16 sP [8][16][136];   // wave-private P tiles

    const int t    = threadIdx.x;
    const int wv   = t >> 5;
    const int lane = t & 31;
    const int half = lane >> 4;
    const int ll   = lane & 15;
    const int row  = t >> 1;          // staging row 0..127
    const int seg  = (t & 1) * 32;    // element offset 0/32

    // ---- async-stage Q tile into sK, read persistent Q fragments -----------
    {
        const __bf16* g = Qh + ((size_t)qt * 128 + row) * D_HEAD + seg;
        async_load_b128(&sK[row][seg + 0],  g + 0);
        async_load_b128(&sK[row][seg + 8],  g + 8);
        async_load_b128(&sK[row][seg + 16], g + 16);
        async_load_b128(&sK[row][seg + 24], g + 24);
    }
    wait_async0();
    __syncthreads();
    bf16x16 qf[2];
#pragma unroll
    for (int c = 0; c < 2; ++c)
        qf[c] = frag_a(&sK[wv * 16 + ll][c * 32], half);
    wait_ds0();            // Q frags landed before anyone overwrites sK
    __syncthreads();

    float mrun[8], lrun[8];
    f32x8 o[4];
#pragma unroll
    for (int e = 0; e < 8; ++e) { mrun[e] = -1e30f; lrun[e] = 0.0f; }
#pragma unroll
    for (int dt = 0; dt < 4; ++dt)
#pragma unroll
        for (int e = 0; e < 8; ++e) o[dt][e] = 0.0f;

    const float scale = 0.125f;  // 1/sqrt(64)

    for (int kc = 0; kc < SEQ_L; kc += 128) {
        {   // async-stage K chunk [128][64]
            const __bf16* g = Kh + (size_t)(kc + row) * D_HEAD + seg;
            async_load_b128(&sK[row][seg + 0],  g + 0);
            async_load_b128(&sK[row][seg + 8],  g + 8);
            async_load_b128(&sK[row][seg + 16], g + 16);
            async_load_b128(&sK[row][seg + 24], g + 24);
        }
        {   // stage V chunk transposed -> sVt[d][key] (manual transpose)
            const __bf16* g = Vh + (size_t)(kc + row) * D_HEAD + seg;
            bf16x8 v0 = *(const bf16x8*)(g + 0);
            bf16x8 v1 = *(const bf16x8*)(g + 8);
            bf16x8 v2 = *(const bf16x8*)(g + 16);
            bf16x8 v3 = *(const bf16x8*)(g + 24);
            __bf16 tmp[32];
            *(bf16x8*)&tmp[0]  = v0; *(bf16x8*)&tmp[8]  = v1;
            *(bf16x8*)&tmp[16] = v2; *(bf16x8*)&tmp[24] = v3;
#pragma unroll
            for (int e = 0; e < 32; ++e) sVt[seg + e][row] = tmp[e];
        }
        wait_async0();
        __syncthreads();

        // ---- S = (Q * K^T) * scale : 8 tiles of 16x16 ----------------------
        f32x8 s[8];
#pragma unroll
        for (int nt = 0; nt < 8; ++nt) {
            f32x8 accs;
#pragma unroll
            for (int e = 0; e < 8; ++e) accs[e] = 0.0f;
#pragma unroll
            for (int c = 0; c < 2; ++c) {
                bf16x16 kfr = frag_b(&sK[nt * 16 + ll][c * 32], half);
                accs = wmma_bf16(qf[c], kfr, accs);
            }
#pragma unroll
            for (int e = 0; e < 8; ++e) s[nt][e] = accs[e] * scale;
        }

        // ---- online softmax (rows across 16-lane halves) -------------------
        float mr[8];
#pragma unroll
        for (int e = 0; e < 8; ++e) {
            mr[e] = s[0][e];
#pragma unroll
            for (int nt = 1; nt < 8; ++nt) mr[e] = fmaxf(mr[e], s[nt][e]);
        }
#pragma unroll
        for (int off = 1; off < 16; off <<= 1)
#pragma unroll
            for (int e = 0; e < 8; ++e)
                mr[e] = fmaxf(mr[e], __shfl_xor(mr[e], off, 16));

        float mnew[8], alpha[8], rs[8];
#pragma unroll
        for (int e = 0; e < 8; ++e) {
            mnew[e]  = fmaxf(mrun[e], mr[e]);
            alpha[e] = __expf(mrun[e] - mnew[e]);
            rs[e]    = 0.0f;
        }
#pragma unroll
        for (int nt = 0; nt < 8; ++nt)
#pragma unroll
            for (int e = 0; e < 8; ++e) {
                float p = __expf(s[nt][e] - mnew[e]);
                s[nt][e] = p;
                rs[e] += p;
            }
#pragma unroll
        for (int off = 1; off < 16; off <<= 1)
#pragma unroll
            for (int e = 0; e < 8; ++e)
                rs[e] += __shfl_xor(rs[e], off, 16);
#pragma unroll
        for (int e = 0; e < 8; ++e) {
            lrun[e] = lrun[e] * alpha[e] + rs[e];
            mrun[e] = mnew[e];
        }
#pragma unroll
        for (int dt = 0; dt < 4; ++dt)
#pragma unroll
            for (int e = 0; e < 8; ++e) o[dt][e] *= alpha[e];

        // ---- P (C-layout) -> wave-private LDS -> A-layout fragments --------
#pragma unroll
        for (int nt = 0; nt < 8; ++nt)
#pragma unroll
            for (int e = 0; e < 8; ++e)
                sP[wv][half ? 8 + e : e][nt * 16 + ll] = (__bf16)s[nt][e];
        wait_ds0();  // same-wave LDS RAW

        // ---- O += P * V ----------------------------------------------------
#pragma unroll
        for (int c = 0; c < 4; ++c) {
            bf16x16 pf = frag_a(&sP[wv][ll][c * 32], half);
#pragma unroll
            for (int dt = 0; dt < 4; ++dt) {
                bf16x16 vfr = frag_b(&sVt[dt * 16 + ll][c * 32], half);
                o[dt] = wmma_bf16(pf, vfr, o[dt]);
            }
        }
        __syncthreads();   // everyone done reading sK/sVt before restage
    }

    // ---- epilogue: O / l -> bf16 attn buffer [B, L, D] ---------------------
#pragma unroll
    for (int dt = 0; dt < 4; ++dt)
#pragma unroll
        for (int e = 0; e < 8; ++e) {
            float val = o[dt][e] / lrun[e];
            int rowe = half ? 8 + e : e;
            size_t l = (size_t)qt * 128 + wv * 16 + rowe;
            O[((size_t)b * SEQ_L + l) * D_MODEL + h * D_HEAD + dt * 16 + ll] =
                (__bf16)val;
        }
}

// ---------------------------------------------------------------------------
// Kernel 3: output projection.  out[m,n] = sum_k A[m,k] * Wo[n,k] -> f32.
// Same 64x64-wave-tile double-buffered GEMM; dense f32 row-major epilogue.
// ---------------------------------------------------------------------------
__global__ __launch_bounds__(256)
void mha_out_proj_kernel(const __bf16* __restrict__ A,
                         const __bf16* __restrict__ Wo,
                         float* __restrict__ out) {
    __shared__ __align__(16) __bf16 sA[2][128][40];
    __shared__ __align__(16) __bf16 sB[2][256][40];

    const int t    = threadIdx.x;
    const int arow = t >> 1;
    const int kseg = (t & 1) << 4;
    const int wv   = t >> 5;
    const int wm   = wv >> 2;
    const int wn   = wv & 3;
    const int lane = t & 31;
    const int half = lane >> 4;
    const int ll   = lane & 15;

    const size_t mbase = (size_t)blockIdx.x * 128;
    const int    nbase = blockIdx.y * 256;

    f32x8 acc[4][4];
#pragma unroll
    for (int i = 0; i < 4; ++i)
#pragma unroll
        for (int j = 0; j < 4; ++j)
#pragma unroll
            for (int e = 0; e < 8; ++e) acc[i][j][e] = 0.0f;

    const __bf16* gA = A  + (mbase + arow) * GEMM_K + kseg;
    const __bf16* gB = Wo + (size_t)(nbase + t) * GEMM_K;

    auto stage = [&](int buf, int k0) {
        const __bf16* pA = gA + k0;
        async_load_b128(&sA[buf][arow][kseg],     pA);
        async_load_b128(&sA[buf][arow][kseg + 8], pA + 8);
        const __bf16* pB = gB + k0;
        async_load_b128(&sB[buf][t][0],  pB);
        async_load_b128(&sB[buf][t][8],  pB + 8);
        async_load_b128(&sB[buf][t][16], pB + 16);
        async_load_b128(&sB[buf][t][24], pB + 24);
    };
    auto compute = [&](int buf) {
        bf16x16 af[4];
#pragma unroll
        for (int i = 0; i < 4; ++i)
            af[i] = frag_a(&sA[buf][wm * 64 + i * 16 + ll][0], half);
#pragma unroll
        for (int j = 0; j < 4; ++j) {
            bf16x16 bfr = frag_b(&sB[buf][wn * 64 + j * 16 + ll][0], half);
#pragma unroll
            for (int i = 0; i < 4; ++i)
                acc[i][j] = wmma_bf16(af[i], bfr, acc[i][j]);
        }
    };

    stage(0, 0);
    int cur = 0;
    for (int k0 = 0; k0 < GEMM_K - 32; k0 += 32) {
        wait_async0();
        __syncthreads();
        stage(cur ^ 1, k0 + 32);
        compute(cur);
        cur ^= 1;
    }
    wait_async0();
    __syncthreads();
    compute(cur);

#pragma unroll
    for (int i = 0; i < 4; ++i)
#pragma unroll
        for (int j = 0; j < 4; ++j)
#pragma unroll
            for (int e = 0; e < 8; ++e) {
                size_t m = mbase + wm * 64 + i * 16 + (half ? 8 + e : e);
                int    n = nbase + wn * 64 + j * 16 + ll;
                out[m * D_MODEL + n] = acc[i][j][e];
            }
}

// ---------------------------------------------------------------------------
extern "C" void kernel_launch(void* const* d_in, const int* in_sizes, int n_in,
                              void* d_out, int out_size, void* d_ws, size_t ws_size,
                              hipStream_t stream) {
    (void)in_sizes; (void)n_in; (void)out_size; (void)ws_size;
    const float* X  = (const float*)d_in[0];
    const float* Wq = (const float*)d_in[1];
    const float* Wk = (const float*)d_in[2];
    const float* Wv = (const float*)d_in[3];
    const float* Wo = (const float*)d_in[4];
    float* out = (float*)d_out;

    const size_t E  = (size_t)BATCH * SEQ_L * D_MODEL;   // 4 Mi elements
    const size_t WE = (size_t)D_MODEL * D_MODEL;         // 256 Ki elements

    __bf16* xb  = (__bf16*)d_ws;
    __bf16* wqb = xb  + E;
    __bf16* wkb = wqb + WE;
    __bf16* wvb = wkb + WE;
    __bf16* wob = wvb + WE;
    __bf16* qb  = wob + WE;
    __bf16* kb  = qb + E;
    __bf16* vb  = kb + E;
    __bf16* ab  = vb + E;

    // 0) one-shot f32 -> bf16 conversion (X + 4 weight matrices)
    mha_cvt_bf16_kernel<<<dim3((unsigned)(E / 2048)), 256, 0, stream>>>(X, xb);
    mha_cvt_bf16_kernel<<<dim3((unsigned)(WE / 2048)), 256, 0, stream>>>(Wq, wqb);
    mha_cvt_bf16_kernel<<<dim3((unsigned)(WE / 2048)), 256, 0, stream>>>(Wk, wkb);
    mha_cvt_bf16_kernel<<<dim3((unsigned)(WE / 2048)), 256, 0, stream>>>(Wv, wvb);
    mha_cvt_bf16_kernel<<<dim3((unsigned)(WE / 2048)), 256, 0, stream>>>(Wo, wob);

    // 1) fused QKV projection: grid.z selects Wq/Wk/Wv
    mha_qkv_proj_kernel<<<dim3(64, 2, 3), 256, 0, stream>>>(
        xb, wqb, wkb, wvb, qb, kb, vb);

    // 2) flash attention over 128-query tiles per (b,h)
    mha_flash_attn_kernel<<<dim3(SEQ_L / 128, N_HEADS, BATCH), 256, 0, stream>>>(
        qb, kb, vb, ab);

    // 3) output projection -> f32
    mha_out_proj_kernel<<<dim3(64, 2), 256, 0, stream>>>(ab, wob, out);
}
